// ACT_basic_64733747085936
// MI455X (gfx1250) — compile-verified, matrix operand order checked by reading, not run on hardware
//
#include <hip/hip_runtime.h>
#include <hip/hip_bf16.h>
#include <cstddef>
#include <cstdint>

// ---------------------------------------------------------------------------
// CDNA5 (gfx1250) ACT layer: 11 hops of (encode + halting dot + FFN GEMMs).
// GEMMs: v_wmma_f32_16x16x32_bf16 (wave32 WMMA), double-buffered LDS tiles
// filled with gfx1250 async global->LDS copies (ASYNCcnt).
// ---------------------------------------------------------------------------

typedef __attribute__((ext_vector_type(16))) __bf16 v16bf;
typedef __attribute__((ext_vector_type(8)))  __bf16 v8bf;
typedef __attribute__((ext_vector_type(8)))  float  v8f;

#define B_   8
#define L_   1024
#define D_   1024
#define DFF_ 4096
#define ROWS (B_ * L_)        // 8192
#define MAX_HOP 11
#define LDS_STRIDE 40         // bf16 elems per row: 80B = 20 banks -> conflict-free b128

// --- gfx1250 async copy helpers --------------------------------------------
// Builtin parameter types (from hipcc diagnostic): (int4 AS1*, int4 AS3*, Ii, Ii)
typedef int v4i_ __attribute__((vector_size(16)));
typedef __attribute__((address_space(1))) v4i_ gv4i_t;
typedef __attribute__((address_space(3))) v4i_ lv4i_t;

__device__ __forceinline__ void async_b128x2(uint32_t lds_addr, int voff,
                                             const __bf16* sbase) {
#if __has_builtin(__builtin_amdgcn_global_load_async_to_lds_b128)
    gv4i_t* g = (gv4i_t*)((const char*)sbase + voff);
    lv4i_t* l = (lv4i_t*)(uintptr_t)lds_addr;
    __builtin_amdgcn_global_load_async_to_lds_b128(g, l, 0, 0);
    __builtin_amdgcn_global_load_async_to_lds_b128(g, l, 16, 0);
#else
    asm volatile("global_load_async_to_lds_b128 %0, %1, %2"
                 :: "v"(lds_addr), "v"(voff), "s"(sbase) : "memory");
    asm volatile("global_load_async_to_lds_b128 %0, %1, %2 offset:16"
                 :: "v"(lds_addr), "v"(voff), "s"(sbase) : "memory");
#endif
}

__device__ __forceinline__ void wait_async_le4() {
#if __has_builtin(__builtin_amdgcn_s_wait_asynccnt)
    __builtin_amdgcn_s_wait_asynccnt(4);
#else
    asm volatile("s_wait_asynccnt 0x4" ::: "memory");
#endif
}
__device__ __forceinline__ void wait_async_0() {
#if __has_builtin(__builtin_amdgcn_s_wait_asynccnt)
    __builtin_amdgcn_s_wait_asynccnt(0);
#else
    asm volatile("s_wait_asynccnt 0x0" ::: "memory");
#endif
}

// ---------------------------------------------------------------------------
// utility kernels
// ---------------------------------------------------------------------------
__global__ __launch_bounds__(256) void zero_f32(float* __restrict__ p, int n) {
    int i = blockIdx.x * 256 + threadIdx.x;
    if (i < n) p[i] = 0.0f;
}

// W[R][C] (fp32, row-major) -> WT[C][R] (bf16, row-major)
__global__ __launch_bounds__(256) void cvt_transpose(const float* __restrict__ W,
                                                     __bf16* __restrict__ WT,
                                                     int R, int C) {
    size_t i = (size_t)blockIdx.x * 256 + threadIdx.x;
    if (i < (size_t)R * C) {
        int r = (int)(i / C);
        int c = (int)(i % C);
        WT[(size_t)c * R + r] = (__bf16)W[i];
    }
}

// ---------------------------------------------------------------------------
// per-hop: s = st + time_enc(l) + pos_enc(hop); p = sigmoid(s.Wp + bp);
// halting-state update; write s as bf16 for the GEMMs.
// ---------------------------------------------------------------------------
__global__ __launch_bounds__(256)
void act_step(const float* __restrict__ st, const float* __restrict__ Wp,
              const float* __restrict__ bp,
              float* __restrict__ hp, float* __restrict__ uw,
              float* __restrict__ rem, float* __restrict__ nu,
              __bf16* __restrict__ sb, int hop) {
    __shared__ float red[8];
    const int row = blockIdx.x;
    const int l   = row & (L_ - 1);
    const int tid = threadIdx.x;
    const float log_inc = 9.210340371976184f / 511.0f;   // log(1e4)/(512-1)

    const float* srow = st + (size_t)row * D_;
    __bf16* sbrow = sb + (size_t)row * D_;

    float sum = 0.0f;
#pragma unroll
    for (int q = 0; q < 4; ++q) {
        const int c    = tid * 4 + q;
        const int half = c >> 9;          // 0: sin-half, 1: cos-half
        const int idx  = c & 511;
        const float inv = __expf(-log_inc * (float)idx);
        const float te  = half ? __cosf((float)l   * inv) : __sinf((float)l   * inv);
        const float pe  = half ? __cosf((float)hop * inv) : __sinf((float)hop * inv);
        const float sv  = srow[c] + te + pe;
        sum += sv * Wp[c];
        sbrow[c] = (__bf16)sv;
    }
#pragma unroll
    for (int off = 16; off > 0; off >>= 1) sum += __shfl_down(sum, off, 32);
    if ((tid & 31) == 0) red[tid >> 5] = sum;
    __syncthreads();
    if (tid == 0) {
        float d = bp[0];
#pragma unroll
        for (int i = 0; i < 8; ++i) d += red[i];
        const float p = 1.0f / (1.0f + __expf(-d));
        float h = hp[row], r = rem[row], n = nu[row];
        const float sr  = (h < 1.0f) ? 1.0f : 0.0f;
        const float acc = h + p * sr;
        const float nh  = ((acc > 0.9f) ? 1.0f : 0.0f) * sr;
        const float sr2 = ((acc <= 0.9f) ? 1.0f : 0.0f) * sr;
        float hn = h + p * sr2;
        const float rn = r + nh * (1.0f - hn);
        hn = hn + nh * rn;
        hp[row]  = hn;
        rem[row] = rn;
        nu[row]  = n + sr2 + nh;
        uw[row]  = p * sr2 + nh * rn;
    }
}

// ---------------------------------------------------------------------------
// WMMA GEMM: C = A[M,K](bf16) x Bt[N,K](bf16, pre-transposed) + bias
// EPI==0: out_bf16 = bf16(relu(C))                 (FFN layer 1)
// EPI==1: out_f32 = C; prev = C*uw + prev*(1-uw)   (FFN layer 2 + ACT blend)
// Block tile 128x128; 8 waves, each 64x32 (4x2 of 16x16x32 WMMA).
// Double-buffered LDS, async global->LDS fills.
// ---------------------------------------------------------------------------
template <int EPI>
__global__ __launch_bounds__(256)
void gemm_wmma(const __bf16* __restrict__ A, const __bf16* __restrict__ Bt,
               const float* __restrict__ bias, int M, int N, int K,
               float* __restrict__ out_f32, __bf16* __restrict__ out_bf16,
               const float* __restrict__ uw, float* __restrict__ prev) {
    __shared__ __bf16 lA[2][128 * LDS_STRIDE];
    __shared__ __bf16 lB[2][128 * LDS_STRIDE];

    const int tid  = threadIdx.x;
    const int lane = tid & 31;
    const int wid  = tid >> 5;
    const int wm   = wid >> 2;        // 0..1 : 64-row slab
    const int wn   = wid & 3;         // 0..3 : 32-col slab
    const int l16  = lane & 15;
    const int lsel = lane >> 4;       // K-half select per ISA A/B layout

    const int m0 = blockIdx.x * 128;
    const int n0 = blockIdx.y * 128;

    const int lrow  = tid >> 1;       // 0..127 : tile row this thread fills
    const int lhalf = tid & 1;        // 16-elem half of the K=32 slice

    // async-fill bookkeeping: per-thread LDS byte addresses + global offsets
    uint32_t dstA[2], dstB[2];
#pragma unroll
    for (int b = 0; b < 2; ++b) {
        dstA[b] = (uint32_t)(uintptr_t)&lA[b][lrow * LDS_STRIDE + lhalf * 16];
        dstB[b] = (uint32_t)(uintptr_t)&lB[b][lrow * LDS_STRIDE + lhalf * 16];
    }
    const int voffA0 = (int)(((size_t)(m0 + lrow) * K + lhalf * 16) * 2);
    const int voffB0 = (int)(((size_t)(n0 + lrow) * K + lhalf * 16) * 2);

    v8f c[4][2];
#pragma unroll
    for (int i = 0; i < 4; ++i)
#pragma unroll
        for (int j = 0; j < 2; ++j)
#pragma unroll
            for (int e = 0; e < 8; ++e) c[i][j][e] = 0.0f;

    const int nk = K >> 5;            // K / 32 steps

    // prologue: stage tile 0 into buffer 0
    async_b128x2(dstA[0], voffA0, A);
    async_b128x2(dstB[0], voffB0, Bt);

    for (int it = 0; it < nk; ++it) {
        const int cur = it & 1;
        if (it + 1 < nk) {            // prefetch tile it+1 into the other buffer
            const int kb = (it + 1) << 6;   // *32 elems * 2 bytes
            async_b128x2(dstA[cur ^ 1], voffA0 + kb, A);
            async_b128x2(dstB[cur ^ 1], voffB0 + kb, Bt);
            wait_async_le4();         // only the newest stage (4 ops) in flight
        } else {
            wait_async_0();
        }
        __syncthreads();              // tile `it` visible to all waves

        v16bf bfrag[2];
#pragma unroll
        for (int j = 0; j < 2; ++j) {
            const __bf16* p = &lB[cur][(wn * 32 + j * 16 + l16) * LDS_STRIDE];
            v8bf lo = *(const v8bf*)(p + 8 * lsel);        // K 0-7  / 8-15
            v8bf hi = *(const v8bf*)(p + 16 + 8 * lsel);   // K 16-23 / 24-31
            bfrag[j] = __builtin_shufflevector(lo, hi, 0, 1, 2, 3, 4, 5, 6, 7,
                                               8, 9, 10, 11, 12, 13, 14, 15);
        }
        v16bf afrag[4];
#pragma unroll
        for (int i = 0; i < 4; ++i) {
            const __bf16* p = &lA[cur][(wm * 64 + i * 16 + l16) * LDS_STRIDE];
            v8bf lo = *(const v8bf*)(p + 8 * lsel);
            v8bf hi = *(const v8bf*)(p + 16 + 8 * lsel);
            afrag[i] = __builtin_shufflevector(lo, hi, 0, 1, 2, 3, 4, 5, 6, 7,
                                               8, 9, 10, 11, 12, 13, 14, 15);
        }
#pragma unroll
        for (int i = 0; i < 4; ++i)
#pragma unroll
            for (int j = 0; j < 2; ++j)
                c[i][j] = __builtin_amdgcn_wmma_f32_16x16x32_bf16(
                    false, afrag[i], false, bfrag[j],
                    (short)0, c[i][j], false, false);

        __syncthreads();              // everyone done reading buf[cur] before it
                                      // is overwritten by stage it+2
    }

    // epilogue: C/D layout — VGPR v, lane: M = v + 8*(lane>>4), N = lane&15
#pragma unroll
    for (int i = 0; i < 4; ++i) {
#pragma unroll
        for (int j = 0; j < 2; ++j) {
            const int colg = n0 + wn * 32 + j * 16 + l16;
            const float bn = bias[colg];
#pragma unroll
            for (int v = 0; v < 8; ++v) {
                const int rowg = m0 + wm * 64 + i * 16 + v + 8 * lsel;
                float x = c[i][j][v] + bn;
                if (EPI == 0) {
                    x = (x > 0.0f) ? x : 0.0f;
                    out_bf16[(size_t)rowg * N + colg] = (__bf16)x;
                } else {
                    const size_t idx = (size_t)rowg * N + colg;
                    out_f32[idx] = x;                       // next-hop state
                    const float w = uw[rowg];
                    prev[idx] = x * w + prev[idx] * (1.0f - w);
                }
            }
        }
    }
}

// ---------------------------------------------------------------------------
// host-side orchestration (graph-capture safe: only stream launches)
// ---------------------------------------------------------------------------
extern "C" void kernel_launch(void* const* d_in, const int* in_sizes, int n_in,
                              void* d_out, int out_size, void* d_ws, size_t ws_size,
                              hipStream_t stream) {
    const float* state = (const float*)d_in[0];
    const float* Wp    = (const float*)d_in[1];
    const float* bp    = (const float*)d_in[2];
    const float* W1    = (const float*)d_in[3];
    const float* b1    = (const float*)d_in[4];
    const float* W2    = (const float*)d_in[5];
    const float* b2    = (const float*)d_in[6];

    float* out  = (float*)d_out;
    float* prev = out;                                  // [8192][1024]
    float* rem  = out + (size_t)ROWS * D_;              // [8192]
    float* nu   = rem + ROWS;                           // [8192]

    char* ws = (char*)d_ws;
    __bf16* W1T = (__bf16*)(ws);                             // [4096][1024]  8 MB
    __bf16* W2T = (__bf16*)(ws + (size_t)8 * 1024 * 1024);   // [1024][4096]  8 MB
    __bf16* S   = (__bf16*)(ws + (size_t)16 * 1024 * 1024);  // [8192][1024] 16 MB
    __bf16* H   = (__bf16*)(ws + (size_t)32 * 1024 * 1024);  // [8192][4096] 64 MB
    float*  ST  = (float*)(ws + (size_t)96 * 1024 * 1024);   // [8192][1024] 32 MB
    float*  HP  = (float*)(ws + (size_t)128 * 1024 * 1024);  // [8192]
    float*  UW  = HP + ROWS;                                 // [8192]

    {
        const int n_out = ROWS * D_ + 2 * ROWS;
        zero_f32<<<(n_out + 255) / 256, 256, 0, stream>>>(out, n_out);
        zero_f32<<<(2 * ROWS + 255) / 256, 256, 0, stream>>>(HP, 2 * ROWS);
    }
    cvt_transpose<<<(D_ * DFF_) / 256, 256, 0, stream>>>(W1, W1T, D_, DFF_);
    cvt_transpose<<<(DFF_ * D_) / 256, 256, 0, stream>>>(W2, W2T, DFF_, D_);

    for (int hop = 0; hop < MAX_HOP; ++hop) {
        const float* src = (hop == 0) ? state : ST;
        act_step<<<ROWS, 256, 0, stream>>>(src, Wp, bp, HP, UW, rem, nu, S, hop);
        // h = relu(s @ W1 + b1) : M=8192 N=4096 K=1024
        gemm_wmma<0><<<dim3(ROWS / 128, DFF_ / 128), 256, 0, stream>>>(
            S, W1T, b1, ROWS, DFF_, D_, nullptr, H, nullptr, nullptr);
        // s2 = h @ W2 + b2 ; prev = s2*uw + prev*(1-uw) ; ST = s2
        gemm_wmma<1><<<dim3(ROWS / 128, D_ / 128), 256, 0, stream>>>(
            H, W2T, b2, ROWS, D_, DFF_, ST, nullptr, UW, prev);
    }
}
